// TwoTowerModel_71708773974789
// MI455X (gfx1250) — matrix operand order repaired and checked by reading, not compiled, not run on hardware
//
#include <hip/hip_runtime.h>
#include <hip/hip_bf16.h>

// ---------------------------------------------------------------------------
// Two-tower scoring on gfx1250 (MI455X), wave32 + v_wmma_f32_16x16x32_f16.
//
// Pipeline per wave (16 batch rows):
//   gather emb rows (fp32) -> A frags (f16)
//   H  = relu(A @ W1 + b1)   16x128, 16 WMMA, spilled to wave-private LDS
//   Y  = relu(H @ W2 + b2)   16x64,  16 WMMA
//   out = (u . v) / (max(|u|,eps) * max(|v|,eps))   via shfl_xor reductions
//
// Weights are pre-packed (fp32 -> f16 B-fragment layout) into d_ws by a tiny
// pack kernel each launch (deterministic, idempotent): 4 x 16 KB = 64 KB.
// ---------------------------------------------------------------------------

typedef __attribute__((ext_vector_type(16))) _Float16 v16h;
typedef __attribute__((ext_vector_type(8)))  _Float16 v8h;
typedef __attribute__((ext_vector_type(8)))  float    v8f;

#define EMB        64
#define H1         128
#define H2         64
#define ROWS_PER_WAVE 16
#define WAVES_PER_BLOCK 8
#define ROWS_PER_BLOCK (ROWS_PER_WAVE * WAVES_PER_BLOCK)

// Packed fragment sizes (halves): (Ktot/32)*(Ntot/16) frags * 32 lanes * 16 halves
#define W1_PACK_HALVES ((EMB / 32) * (H1 / 16) * 512)   // 2*8*512  = 8192
#define W2_PACK_HALVES ((H1 / 32) * (H2 / 16) * 512)    // 4*4*512  = 8192

// ---------------------------------------------------------------------------
// Pack fp32 weight W[Ktot x Ntot] (row-major, as in the reference: W[k][n])
// into the 16-bit B-matrix (KxN) WMMA fragment layout:
//   fragment (s, t): lane l (g = l/16, n = t*16 + l%16) holds halves
//   j -> element K = s*32 + g*8 + j (j < 8) or s*32 + 16 + g*8 + (j - 8) (j >= 8), column n.
// Stored lane-major so the main kernel reads one contiguous v16h per lane.
// ---------------------------------------------------------------------------
__global__ __launch_bounds__(256)
void tt_pack_weights(const float* __restrict__ W, _Float16* __restrict__ outp,
                     int Ktot, int Ntot) {
  int tid = blockIdx.x * blockDim.x + threadIdx.x;
  int NT = Ntot >> 4;
  int total = (Ktot >> 5) * NT * 512;
  if (tid >= total) return;
  int frag   = tid >> 9;        // which (s,t) fragment
  int within = tid & 511;
  int lane   = within >> 4;
  int j      = within & 15;
  int s = frag / NT;
  int t = frag - s * NT;
  int g = lane >> 4;
  int k = s * 32 + g * 8 + (j < 8 ? j : j + 8);
  int n = t * 16 + (lane & 15);
  outp[tid] = (_Float16)W[k * Ntot + n];
}

__device__ __forceinline__ v8f wmma16(v16h a, v16h b, v8f c) {
  // (neg_a, A, neg_b, B, c_mod, C, reuse_a, reuse_b)
  return __builtin_amdgcn_wmma_f32_16x16x32_f16(false, a, false, b,
                                                (short)0, c, false, false);
}

// ---------------------------------------------------------------------------
// One tower for one 16-row wave tile. Produces Y as 4 D-fragments (f32).
// H: wave-private LDS slice, 16*128 halves (4 KB).
// ---------------------------------------------------------------------------
__device__ __forceinline__ void run_tower(const int*     __restrict__ ids,
                                          const float*   __restrict__ table,
                                          const float*   __restrict__ b1,
                                          const float*   __restrict__ b2,
                                          const _Float16* __restrict__ w1p,
                                          const _Float16* __restrict__ w2p,
                                          _Float16*      __restrict__ H,
                                          int rowBase, int lane, v8f y[4]) {
  const int g  = lane >> 4;   // half-group of the wave
  const int mn = lane & 15;   // A row / B,C,D column within tile

  // ---- A fragments from gathered embedding row (fp32 -> f16) -------------
  // A f16 16x32 layout: lane (g,m) holds K = g*8 + {0..7} and g*8+16+{0..7}.
  const long long id = (long long)ids[rowBase + mn];
  const float4* rp4 = (const float4*)(table + id * EMB + g * 8);
  float4 f0 = rp4[0],  f1 = rp4[1];    // cols g8 + 0..7
  float4 f2 = rp4[4],  f3 = rp4[5];    // cols g8 + 16..23
  float4 f4 = rp4[8],  f5 = rp4[9];    // cols g8 + 32..39
  float4 f6 = rp4[12], f7 = rp4[13];   // cols g8 + 48..55
  v16h a0, a1;
  a0[0]=(_Float16)f0.x; a0[1]=(_Float16)f0.y; a0[2]=(_Float16)f0.z; a0[3]=(_Float16)f0.w;
  a0[4]=(_Float16)f1.x; a0[5]=(_Float16)f1.y; a0[6]=(_Float16)f1.z; a0[7]=(_Float16)f1.w;
  a0[8]=(_Float16)f2.x; a0[9]=(_Float16)f2.y; a0[10]=(_Float16)f2.z; a0[11]=(_Float16)f2.w;
  a0[12]=(_Float16)f3.x; a0[13]=(_Float16)f3.y; a0[14]=(_Float16)f3.z; a0[15]=(_Float16)f3.w;
  a1[0]=(_Float16)f4.x; a1[1]=(_Float16)f4.y; a1[2]=(_Float16)f4.z; a1[3]=(_Float16)f4.w;
  a1[4]=(_Float16)f5.x; a1[5]=(_Float16)f5.y; a1[6]=(_Float16)f5.z; a1[7]=(_Float16)f5.w;
  a1[8]=(_Float16)f6.x; a1[9]=(_Float16)f6.y; a1[10]=(_Float16)f6.z; a1[11]=(_Float16)f6.w;
  a1[12]=(_Float16)f7.x; a1[13]=(_Float16)f7.y; a1[14]=(_Float16)f7.z; a1[15]=(_Float16)f7.w;

  // ---- GEMM1: H[16x128] = relu(A[16x64] @ W1[64x128] + b1) ---------------
#pragma unroll
  for (int t = 0; t < H1 / 16; ++t) {
    float bias = b1[t * 16 + mn];             // lane owns one N column
    v8f acc;
#pragma unroll
    for (int r = 0; r < 8; ++r) acc[r] = bias;
    v16h bA = *(const v16h*)(w1p + ((0 * (H1/16) + t) * 32 + lane) * 16);
    v16h bB = *(const v16h*)(w1p + ((1 * (H1/16) + t) * 32 + lane) * 16);
    acc = wmma16(a0, bA, acc);
    acc = wmma16(a1, bB, acc);
#pragma unroll
    for (int r = 0; r < 8; ++r) {
      float v = acc[r] > 0.0f ? acc[r] : 0.0f;          // ReLU
      H[(r + 8 * g) * H1 + t * 16 + mn] = (_Float16)v;  // D layout -> row-major
    }
  }

  // ---- Reload H as A fragments (wave-private LDS: in-order, no barrier) --
  v16h ha[4];
#pragma unroll
  for (int s = 0; s < 4; ++s) {
    v8h lo = *(const v8h*)(H + mn * H1 + s * 32 + g * 8);
    v8h hi = *(const v8h*)(H + mn * H1 + s * 32 + g * 8 + 16);
#pragma unroll
    for (int c = 0; c < 8; ++c) { ha[s][c] = lo[c]; ha[s][c + 8] = hi[c]; }
  }

  // ---- GEMM2: Y[16x64] = relu(H[16x128] @ W2[128x64] + b2) ---------------
#pragma unroll
  for (int t = 0; t < H2 / 16; ++t) {
    float bias = b2[t * 16 + mn];
    v8f acc;
#pragma unroll
    for (int r = 0; r < 8; ++r) acc[r] = bias;
#pragma unroll
    for (int s = 0; s < 4; ++s) {
      v16h bf = *(const v16h*)(w2p + ((s * (H2/16) + t) * 32 + lane) * 16);
      acc = wmma16(ha[s], bf, acc);
    }
#pragma unroll
    for (int r = 0; r < 8; ++r) acc[r] = acc[r] > 0.0f ? acc[r] : 0.0f;
    y[t] = acc;
  }
}

__global__ __launch_bounds__(256)
void tt_score_kernel(const int*   __restrict__ uids,
                     const int*   __restrict__ iids,
                     const float* __restrict__ utab,
                     const float* __restrict__ itab,
                     const float* __restrict__ ub1,
                     const float* __restrict__ ub2,
                     const float* __restrict__ ib1,
                     const float* __restrict__ ib2,
                     const _Float16* __restrict__ wpack,
                     float* __restrict__ out) {
  __shared__ __align__(32) _Float16 Hlds[WAVES_PER_BLOCK][ROWS_PER_WAVE * H1]; // 32 KB

  const int tid  = threadIdx.x;
  const int wave = tid >> 5;
  const int lane = tid & 31;
  const int g    = lane >> 4;
  const int tile = blockIdx.x * WAVES_PER_BLOCK + wave;
  const int rowBase = tile * ROWS_PER_WAVE;

  _Float16* H = &Hlds[wave][0];

  v8f yu[4], yi[4];
  run_tower(uids, utab, ub1, ub2,
            wpack, wpack + W1_PACK_HALVES,
            H, rowBase, lane, yu);
  run_tower(iids, itab, ib1, ib2,
            wpack + W1_PACK_HALVES + W2_PACK_HALVES,
            wpack + 2 * W1_PACK_HALVES + W2_PACK_HALVES,
            H, rowBase, lane, yi);

  // Epilogue: cosine score per row. Row (r + 8g) lives across the 16 lanes of
  // this wave half (4 tiles x 1 column each); reduce with intra-half shfl_xor.
#pragma unroll
  for (int r = 0; r < 8; ++r) {
    float su = 0.0f, si = 0.0f, sd = 0.0f;
#pragma unroll
    for (int t = 0; t < 4; ++t) {
      float u = yu[t][r], v = yi[t][r];
      su += u * u;
      si += v * v;
      sd += u * v;
    }
#pragma unroll
    for (int m = 1; m <= 8; m <<= 1) {   // stays within each 16-lane half
      su += __shfl_xor(su, m, 32);
      si += __shfl_xor(si, m, 32);
      sd += __shfl_xor(sd, m, 32);
    }
    float nu = fmaxf(sqrtf(su), 1e-12f);
    float nv = fmaxf(sqrtf(si), 1e-12f);
    float res = sd / (nu * nv);
    if ((lane & 15) == 0)                 // lanes 0 / 16 write rows r / 8+r
      out[rowBase + g * 8 + r] = res;
  }
}

extern "C" void kernel_launch(void* const* d_in, const int* in_sizes, int n_in,
                              void* d_out, int out_size, void* d_ws, size_t ws_size,
                              hipStream_t stream) {
  // setup_inputs order:
  // 0 user_ids, 1 item_ids, 2 user_table, 3 item_table,
  // 4 uW1, 5 ub1, 6 uW2, 7 ub2, 8 iW1, 9 ib1, 10 iW2, 11 ib2
  const int*   uids = (const int*)d_in[0];
  const int*   iids = (const int*)d_in[1];
  const float* utab = (const float*)d_in[2];
  const float* itab = (const float*)d_in[3];
  const float* uW1  = (const float*)d_in[4];
  const float* ub1  = (const float*)d_in[5];
  const float* uW2  = (const float*)d_in[6];
  const float* ub2  = (const float*)d_in[7];
  const float* iW1  = (const float*)d_in[8];
  const float* ib1  = (const float*)d_in[9];
  const float* iW2  = (const float*)d_in[10];
  const float* ib2  = (const float*)d_in[11];

  _Float16* wpack = (_Float16*)d_ws;   // 4 x 8192 halves = 64 KB

  // Pack the four weight matrices into WMMA B-fragment layout (f16).
  {
    dim3 blk(256);
    dim3 grdA((W1_PACK_HALVES + 255) / 256);
    dim3 grdB((W2_PACK_HALVES + 255) / 256);
    tt_pack_weights<<<grdA, blk, 0, stream>>>(uW1, wpack,                                   EMB, H1);
    tt_pack_weights<<<grdB, blk, 0, stream>>>(uW2, wpack + W1_PACK_HALVES,                  H1,  H2);
    tt_pack_weights<<<grdA, blk, 0, stream>>>(iW1, wpack + W1_PACK_HALVES + W2_PACK_HALVES, EMB, H1);
    tt_pack_weights<<<grdB, blk, 0, stream>>>(iW2, wpack + 2*W1_PACK_HALVES + W2_PACK_HALVES, H1, H2);
  }

  const int B = in_sizes[0];                    // 524288
  dim3 blk(256);
  dim3 grd(B / ROWS_PER_BLOCK);                 // 4096 blocks, exact cover
  tt_score_kernel<<<grd, blk, 0, stream>>>(uids, iids, utab, itab,
                                           ub1, ub2, ib1, ib2,
                                           wpack, (float*)d_out);
}
